// CrossAttentionK_65601330479495
// MI455X (gfx1250) — compile-verified, matrix operand order checked by reading, not run on hardware
//
#include <hip/hip_runtime.h>

// ---------------------------------------------------------------------------
// MI455X (gfx1250) implementation. Everything matmul-shaped runs through
// v_wmma_f32_16x16x32_bf16 (wave32). ~755 GFLOP vs <1GB HBM traffic at
// 23.3 TB/s -> compute bound -> bf16 WMMA, f32 accumulate.
// GEMM/conv use LDS ping-pong double buffering: global loads for step i+1
// issue before the WMMAs of step i; fragments are batch-loaded so one
// s_wait_dscnt covers an 8-deep v_wmma burst.
// ---------------------------------------------------------------------------

typedef __attribute__((ext_vector_type(16))) __bf16          v16bf;
typedef __attribute__((ext_vector_type(8)))  float           v8f;
typedef __attribute__((ext_vector_type(8)))  unsigned short  u16x8;
typedef __attribute__((ext_vector_type(16))) unsigned short  u16x16;

#define B_SZ   1024
#define NQ     64
#define C_SZ   512
#define HH     8
#define DK     64
#define SLOT   33554432   // B*NQ*C = 1024*64*512 elements per big buffer

// f32 -> bf16 bits, round-to-nearest-even
__device__ __forceinline__ unsigned short f2bf(float f) {
  unsigned u = __builtin_bit_cast(unsigned, f);
  unsigned r = u + 0x7fffu + ((u >> 16) & 1u);
  return (unsigned short)(r >> 16);
}

__device__ __forceinline__ v8f vzero8() {
  v8f z;
#pragma unroll
  for (int i = 0; i < 8; ++i) z[i] = 0.0f;
  return z;
}

// Load a 16x32 A fragment (row=M, K contiguous) or a 32x16 B fragment stored
// as Bt[n][k] from LDS. Documented layout: lane = (m|n)%16 + 16*half, element
// e covers K = (e/8)*16 + half*8 + (e%8) -> two contiguous b128 LDS loads.
__device__ __forceinline__ v16bf load_frag(const unsigned short* base, int row,
                                           int ld, int kbase, int lane) {
  int half = (lane >> 4) & 1;
  const unsigned short* p = base + row * ld + kbase + half * 8;
  u16x8 lo = *(const u16x8*)(p);
  u16x8 hi = *(const u16x8*)(p + 16);
  u16x16 t = __builtin_shufflevector(lo, hi, 0,1,2,3,4,5,6,7,8,9,10,11,12,13,14,15);
  return __builtin_bit_cast(v16bf, t);
}

__device__ __forceinline__ v8f wmma_bf16(v16bf a, v16bf b, v8f c) {
  return __builtin_amdgcn_wmma_f32_16x16x32_bf16(false, a, false, b, (short)0, c,
                                                 false, false);
}

__device__ __forceinline__ void st4bf(unsigned short* p, float4 v) {
  p[0] = f2bf(v.x); p[1] = f2bf(v.y); p[2] = f2bf(v.z); p[3] = f2bf(v.w);
}

// ---------------------------------------------------------------------------
// Generic GEMM:  C[m,n] = sum_k A[m,k] * W[n,k] + bias[n]
// A optionally K-split into (A0|A1) for the concat([x,x2]) case.
// Tile 128x128, 8 waves (4M x 2N), wave tile 32x64, KT=32, LDS ping-pong.
// ---------------------------------------------------------------------------
__global__ __launch_bounds__(256) void gemm_bias_kernel(
    const float* __restrict__ A0, int K0,
    const float* __restrict__ A1, int K1,
    const float* __restrict__ W, const float* __restrict__ bias,
    float* __restrict__ Cm, int N) {
  const int K = K0 + K1;
  __shared__ unsigned short As[2][128 * 32];
  __shared__ unsigned short Bs[2][128 * 32];
  int t = threadIdx.x, lane = t & 31, wave = t >> 5;
  int wm = wave & 3, wn = wave >> 2;
  int mBase = blockIdx.x * 128, nBase = blockIdx.y * 128;
  int ln = lane & 15, half = lane >> 4;

  v8f acc[2][4];
#pragma unroll
  for (int i = 0; i < 2; ++i)
#pragma unroll
    for (int j = 0; j < 4; ++j) acc[i][j] = vzero8();

  float4 ra[4], rb[4];
  // issue global loads for one 128x32 K-tile (A + W) into registers
  auto stage_load = [&](int kb) {
    const float* Asrc; int kloc, lda;
    if (kb < K0) { Asrc = A0; kloc = kb;      lda = K0; }
    else         { Asrc = A1; kloc = kb - K0; lda = K1; }
#pragma unroll
    for (int j = 0; j < 4; ++j) {
      int off = (t + j * 256) * 4;
      int r = off >> 5, c = off & 31;
      ra[j] = *(const float4*)(Asrc + (size_t)(mBase + r) * lda + kloc + c);
      rb[j] = *(const float4*)(W + (size_t)(nBase + r) * K + kb + c);
    }
  };
  auto stage_store = [&](int buf) {
#pragma unroll
    for (int j = 0; j < 4; ++j) {
      int off = (t + j * 256) * 4;
      int r = off >> 5, c = off & 31;
      st4bf(&As[buf][r * 32 + c], ra[j]);
      st4bf(&Bs[buf][r * 32 + c], rb[j]);
    }
  };

  stage_load(0);
  stage_store(0);
  int cur = 0;
  for (int kb = 0; kb < K; kb += 32) {
    __syncthreads();
    bool more = (kb + 32) < K;
    if (more) stage_load(kb + 32);                 // overlap with compute below
    if (kb + 64 < K)                               // speculative far prefetch
      __builtin_prefetch(W + (size_t)(nBase + (t >> 1)) * K + kb + 64, 0, 1);

    // batch all fragments, then an 8-deep WMMA burst
    v16bf a0 = load_frag(As[cur], wm * 32 + ln,      32, 0, lane);
    v16bf a1 = load_frag(As[cur], wm * 32 + 16 + ln, 32, 0, lane);
    v16bf bf[4];
#pragma unroll
    for (int nt = 0; nt < 4; ++nt)
      bf[nt] = load_frag(Bs[cur], wn * 64 + nt * 16 + ln, 32, 0, lane);
#pragma unroll
    for (int nt = 0; nt < 4; ++nt) {
      acc[0][nt] = wmma_bf16(a0, bf[nt], acc[0][nt]);
      acc[1][nt] = wmma_bf16(a1, bf[nt], acc[1][nt]);
    }
    if (more) stage_store(cur ^ 1);                // waits on global loads only
    cur ^= 1;
  }
#pragma unroll
  for (int mt = 0; mt < 2; ++mt)
#pragma unroll
    for (int nt = 0; nt < 4; ++nt) {
      int ncol = nBase + wn * 64 + nt * 16 + ln;
      float bv = bias ? bias[ncol] : 0.0f;
#pragma unroll
      for (int r = 0; r < 8; ++r) {
        int mrow = mBase + wm * 32 + mt * 16 + r + 8 * half;
        Cm[(size_t)mrow * N + ncol] = acc[mt][nt][r] + bv;
      }
    }
}

// ---------------------------------------------------------------------------
// 3x3 SAME conv on (B,512,8,8) as implicit GEMM. Input yT is (b,pos,ch),
// output written transposed: outT[b,pos,outch]. Block: one batch x 128 outch.
// 144 K-steps (9 taps x 16), LDS ping-pong as above.
// ---------------------------------------------------------------------------
__global__ __launch_bounds__(256) void conv3x3_kernel(
    const float* __restrict__ yT, const float* __restrict__ w,
    const float* __restrict__ bias, float* __restrict__ outT) {
  __shared__ unsigned short As[2][64 * 32];
  __shared__ unsigned short Bs[2][128 * 32];
  int b = blockIdx.y, nBase = blockIdx.x * 128;
  int t = threadIdx.x, lane = t & 31, wave = t >> 5;
  int wm = wave & 1, wn = wave >> 1;
  int ln = lane & 15, half = lane >> 4;
  size_t base = (size_t)b * (NQ * C_SZ);

  v8f acc[2][2];
#pragma unroll
  for (int i = 0; i < 2; ++i)
#pragma unroll
    for (int j = 0; j < 2; ++j) acc[i][j] = vzero8();

  float4 rca[2];
  float  rcb[16];
  auto stage_load = [&](int it) {            // it = tap*16 + kc
    int tap = it >> 4, kb = (it & 15) * 32;
    int kh = tap / 3 - 1, kw = tap % 3 - 1;
#pragma unroll
    for (int j = 0; j < 2; ++j) {            // 64x32 shifted input tile
      int off = (t + j * 256) * 4;
      int p = off >> 5, c = off & 31;
      int py = p >> 3, px = p & 7;
      int sy = py + kh, sx = px + kw;
      float4 v; v.x = v.y = v.z = v.w = 0.0f;
      if (sy >= 0 && sy < 8 && sx >= 0 && sx < 8)
        v = *(const float4*)(yT + base + (size_t)(sy * 8 + sx) * C_SZ + kb + c);
      rca[j] = v;
    }
#pragma unroll
    for (int j = 0; j < 16; ++j) {           // 128x32 weight tile (stride-9)
      int idx = t + j * 256;
      int o = idx >> 5, c = idx & 31;
      rcb[j] = w[(size_t)(nBase + o) * (C_SZ * 9) + (size_t)(kb + c) * 9 + tap];
    }
  };
  auto stage_store = [&](int buf) {
#pragma unroll
    for (int j = 0; j < 2; ++j) {
      int off = (t + j * 256) * 4;
      int p = off >> 5, c = off & 31;
      st4bf(&As[buf][p * 32 + c], rca[j]);
    }
#pragma unroll
    for (int j = 0; j < 16; ++j) {
      int idx = t + j * 256;
      int o = idx >> 5, c = idx & 31;
      Bs[buf][o * 32 + c] = f2bf(rcb[j]);
    }
  };

  stage_load(0);
  stage_store(0);
  int cur = 0;
  for (int it = 0; it < 144; ++it) {
    __syncthreads();
    bool more = (it + 1) < 144;
    if (more) stage_load(it + 1);
    v16bf a0 = load_frag(As[cur], wm * 32 + ln,      32, 0, lane);
    v16bf a1 = load_frag(As[cur], wm * 32 + 16 + ln, 32, 0, lane);
    v16bf bf[2];
#pragma unroll
    for (int nt = 0; nt < 2; ++nt)
      bf[nt] = load_frag(Bs[cur], wn * 32 + nt * 16 + ln, 32, 0, lane);
#pragma unroll
    for (int nt = 0; nt < 2; ++nt) {
      acc[0][nt] = wmma_bf16(a0, bf[nt], acc[0][nt]);
      acc[1][nt] = wmma_bf16(a1, bf[nt], acc[1][nt]);
    }
    if (more) stage_store(cur ^ 1);
    cur ^= 1;
  }
#pragma unroll
  for (int mt = 0; mt < 2; ++mt)
#pragma unroll
    for (int nt = 0; nt < 2; ++nt) {
      int ncol = nBase + wn * 32 + nt * 16 + ln;
      float bv = bias[ncol];
#pragma unroll
      for (int r = 0; r < 8; ++r) {
        int m = wm * 32 + mt * 16 + r + 8 * half;     // position
        outT[base + (size_t)m * C_SZ + ncol] = acc[mt][nt][r] + bv;
      }
    }
}

// ---------------------------------------------------------------------------
// Fused per-(b,h) attention: S = Q Kt / 8 [*aw, mask -> -inf], softmax rows,
// O = P V.  mode 0: out = meanT + gamma*O + resid (complex attention)
// mode 1: plain store (cross attention).  4 waves, each owns 16 rows.
// ---------------------------------------------------------------------------
__global__ __launch_bounds__(128) void attn_kernel(
    const float* __restrict__ Qb, const float* __restrict__ Kb,
    const float* __restrict__ Vb, const float* __restrict__ aw,
    const unsigned char* __restrict__ mask, const float* __restrict__ meanT,
    const float* __restrict__ resid, const float* __restrict__ gammaPtr,
    float* __restrict__ outB, int mode) {
  __shared__ unsigned short Qs[64 * 64];
  __shared__ unsigned short Ks[64 * 64];   // Bt[j][d] = K[j,d] (natural layout)
  __shared__ unsigned short Vt[64 * 64];   // Bt[d][j] = V[j,d] (transposed)
  __shared__ unsigned short Ps[64 * 64];
  int h = blockIdx.x, b = blockIdx.y;
  int t = threadIdx.x, lane = t & 31, wave = t >> 5;
  int ln = lane & 15, half = lane >> 4;
  size_t base = (size_t)b * (NQ * C_SZ);

#pragma unroll
  for (int j = 0; j < 8; ++j) {            // Q and K: 64x64 f32 each
    int off = (t + j * 128) * 4;
    int i = off >> 6, c = off & 63;
    float4 q4 = *(const float4*)(Qb + base + (size_t)i * C_SZ + h * DK + c);
    st4bf(&Qs[i * 64 + c], q4);
    float4 k4 = *(const float4*)(Kb + base + (size_t)i * C_SZ + h * DK + c);
    st4bf(&Ks[i * 64 + c], k4);
  }
#pragma unroll
  for (int j = 0; j < 32; ++j) {           // V transposed into Vt[d][j]
    int idx = t + j * 128;
    int vj = idx >> 6, d = idx & 63;
    Vt[d * 64 + vj] = f2bf(Vb[base + (size_t)vj * C_SZ + h * DK + d]);
  }
  __syncthreads();

  int iBase = wave * 16;
  v8f s[4];
#pragma unroll
  for (int jt = 0; jt < 4; ++jt) s[jt] = vzero8();
  {
    v16bf a0 = load_frag(Qs, iBase + ln, 64, 0,  lane);
    v16bf a1 = load_frag(Qs, iBase + ln, 64, 32, lane);
    v16bf kf[8];
#pragma unroll
    for (int jt = 0; jt < 4; ++jt) {
      kf[jt]     = load_frag(Ks, jt * 16 + ln, 64, 0,  lane);
      kf[4 + jt] = load_frag(Ks, jt * 16 + ln, 64, 32, lane);
    }
#pragma unroll
    for (int jt = 0; jt < 4; ++jt) {
      s[jt] = wmma_bf16(a0, kf[jt],     s[jt]);
      s[jt] = wmma_bf16(a1, kf[4 + jt], s[jt]);
    }
  }
  const float scale = 0.125f;              // 1/sqrt(64)
#pragma unroll
  for (int jt = 0; jt < 4; ++jt)
#pragma unroll
    for (int r = 0; r < 8; ++r) {
      float v = s[jt][r] * scale;
      if (mode == 1) {
        int i = iBase + r + 8 * half;
        int jc = jt * 16 + ln;
        size_t aidx = (((size_t)b * HH + h) * NQ + i) * NQ + jc;
        v *= aw[aidx];
        if (mask[aidx]) v = -__builtin_huge_valf();
      }
      s[jt][r] = v;
    }
  // softmax across each row (4 jt tiles x 16 lanes within half-wave)
#pragma unroll
  for (int r = 0; r < 8; ++r) {
    float m = fmaxf(fmaxf(s[0][r], s[1][r]), fmaxf(s[2][r], s[3][r]));
    for (int o = 8; o >= 1; o >>= 1) m = fmaxf(m, __shfl_xor(m, o, 16));
    float sum = 0.0f;
#pragma unroll
    for (int jt = 0; jt < 4; ++jt) {
      float e = __expf(s[jt][r] - m);
      s[jt][r] = e; sum += e;
    }
    for (int o = 8; o >= 1; o >>= 1) sum += __shfl_xor(sum, o, 16);
    float inv = 1.0f / sum;
#pragma unroll
    for (int jt = 0; jt < 4; ++jt) s[jt][r] *= inv;
  }
#pragma unroll
  for (int jt = 0; jt < 4; ++jt)
#pragma unroll
    for (int r = 0; r < 8; ++r) {
      int i = iBase + r + 8 * half;
      Ps[i * 64 + jt * 16 + ln] = f2bf(s[jt][r]);
    }
  __syncthreads();

  v8f o[4];
#pragma unroll
  for (int dt = 0; dt < 4; ++dt) o[dt] = vzero8();
  {
    v16bf a0 = load_frag(Ps, iBase + ln, 64, 0,  lane);
    v16bf a1 = load_frag(Ps, iBase + ln, 64, 32, lane);
    v16bf vf[8];
#pragma unroll
    for (int dt = 0; dt < 4; ++dt) {
      vf[dt]     = load_frag(Vt, dt * 16 + ln, 64, 0,  lane);
      vf[4 + dt] = load_frag(Vt, dt * 16 + ln, 64, 32, lane);
    }
#pragma unroll
    for (int dt = 0; dt < 4; ++dt) {
      o[dt] = wmma_bf16(a0, vf[dt],     o[dt]);
      o[dt] = wmma_bf16(a1, vf[4 + dt], o[dt]);
    }
  }
  float gamma = (mode == 0) ? gammaPtr[0] : 0.0f;
#pragma unroll
  for (int dt = 0; dt < 4; ++dt)
#pragma unroll
    for (int r = 0; r < 8; ++r) {
      int i = iBase + r + 8 * half;
      int c = h * DK + dt * 16 + ln;
      size_t oidx = base + (size_t)i * C_SZ + c;
      float val = o[dt][r];
      if (mode == 0) val = meanT[i * C_SZ + c] + gamma * val + resid[oidx];
      outB[oidx] = val;
    }
}

// in: (B,64,512) raw == mq (B,512,8,8). out[b,p,c] = in-as-mq[b,c,p]
__global__ __launch_bounds__(256) void transpose_kernel(
    const float* __restrict__ in, float* __restrict__ out) {
  size_t base = (size_t)blockIdx.x * (NQ * C_SZ);
#pragma unroll 4
  for (int j = 0; j < 128; ++j) {
    int idx = threadIdx.x + j * 256;
    int p = idx >> 9, c = idx & 511;
    out[base + idx] = in[base + c * NQ + p];
  }
}

// meanT[p,c] = mean_b mqT[b,p,c]
__global__ __launch_bounds__(256) void mean_kernel(const float* __restrict__ mqT,
                                                   float* __restrict__ meanT) {
  int idx = blockIdx.x * 256 + threadIdx.x;   // 32768 = 64*512
  float s = 0.0f;
  for (int b = 0; b < B_SZ; ++b) s += mqT[(size_t)b * (NQ * C_SZ) + idx];
  meanT[idx] = s * (1.0f / (float)B_SZ);
}

// ---------------------------------------------------------------------------
extern "C" void kernel_launch(void* const* d_in, const int* in_sizes, int n_in,
                              void* d_out, int out_size, void* d_ws, size_t ws_size,
                              hipStream_t stream) {
  (void)in_sizes; (void)n_in; (void)out_size; (void)ws_size;
  const float* x        = (const float*)d_in[0];
  const float* x2       = (const float*)d_in[1];
  const float* aw       = (const float*)d_in[2];
  const unsigned char* mask = (const unsigned char*)d_in[3];
  const float* mixq_w   = (const float*)d_in[4];
  const float* mixq_b   = (const float*)d_in[5];
  const float* mixqp_w  = (const float*)d_in[6];
  const float* mixqp_b  = (const float*)d_in[7];
  const float* ca_qw    = (const float*)d_in[8];
  const float* ca_qb    = (const float*)d_in[9];
  const float* ca_kw    = (const float*)d_in[10];
  const float* ca_kb    = (const float*)d_in[11];
  const float* ca_vw    = (const float*)d_in[12];
  const float* ca_vb    = (const float*)d_in[13];
  const float* ca_gamma = (const float*)d_in[14];
  const float* cba_cw   = (const float*)d_in[15];
  const float* cba_cb   = (const float*)d_in[16];
  const float* rgbk_w   = (const float*)d_in[17];
  const float* rgbk_b   = (const float*)d_in[18];
  const float* rgbv_w   = (const float*)d_in[19];
  const float* rgbv_b   = (const float*)d_in[20];
  const float* rgbo_w   = (const float*)d_in[21];
  const float* rgbo_b   = (const float*)d_in[22];
  const float* infk_w   = (const float*)d_in[23];
  const float* infk_b   = (const float*)d_in[24];
  const float* infv_w   = (const float*)d_in[25];
  const float* infv_b   = (const float*)d_in[26];
  const float* info_w   = (const float*)d_in[27];
  const float* info_b   = (const float*)d_in[28];

  float* out_rgb = (float*)d_out;
  float* out_inf = (float*)d_out + (size_t)SLOT;

  float* ws = (float*)d_ws;
  float* P0 = ws + 0 * (size_t)SLOT;
  float* P1 = ws + 1 * (size_t)SLOT;
  float* P2 = ws + 2 * (size_t)SLOT;
  float* P3 = ws + 3 * (size_t)SLOT;
  float* P4 = ws + 4 * (size_t)SLOT;
  float* P5 = ws + 5 * (size_t)SLOT;
  float* meanT = ws + 6 * (size_t)SLOT;   // 64*512 floats

  dim3 gGemm(512, 4);          // M=65536/128, N=512/128
  dim3 gConv(4, B_SZ);
  dim3 gAttn(HH, B_SZ);

  // 1. mix_q = [x|x2] @ mixq_w^T + b  -> P0 (raw-view = mq (B,512,8,8))
  gemm_bias_kernel<<<gGemm, 256, 0, stream>>>(x, 512, x2, 512, mixq_w, mixq_b, P0, C_SZ);
  // 2. mqT[b,p,c] = mq[b,c,p] -> P1 ; 3. batch mean -> meanT
  transpose_kernel<<<B_SZ, 256, 0, stream>>>(P0, P1);
  mean_kernel<<<128, 256, 0, stream>>>(P1, meanT);
  // 4-6. complex-attention projections (on mqT) -> P2,P3,P4
  gemm_bias_kernel<<<gGemm, 256, 0, stream>>>(P1, 512, nullptr, 0, ca_qw, ca_qb, P2, C_SZ);
  gemm_bias_kernel<<<gGemm, 256, 0, stream>>>(P1, 512, nullptr, 0, ca_kw, ca_kb, P3, C_SZ);
  gemm_bias_kernel<<<gGemm, 256, 0, stream>>>(P1, 512, nullptr, 0, ca_vw, ca_vb, P4, C_SZ);
  // 7. yT = meanT + gamma*softmax(QK^T/8)V + mqT  -> P5
  attn_kernel<<<gAttn, 128, 0, stream>>>(P2, P3, P4, nullptr, nullptr, meanT, P1,
                                         ca_gamma, P5, 0);
  // 8. 3x3 conv (implicit GEMM), output transposed (b,pos,ch) -> P0
  conv3x3_kernel<<<gConv, 256, 0, stream>>>(P5, cba_cw, cba_cb, P0);
  // 9. q = convT @ mixqp_w^T + b -> P1
  gemm_bias_kernel<<<gGemm, 256, 0, stream>>>(P0, 512, nullptr, 0, mixqp_w, mixqp_b, P1, C_SZ);
  // 10-13. k/v projections
  gemm_bias_kernel<<<gGemm, 256, 0, stream>>>(x,  512, nullptr, 0, rgbk_w, rgbk_b, P2, C_SZ);
  gemm_bias_kernel<<<gGemm, 256, 0, stream>>>(x2, 512, nullptr, 0, infk_w, infk_b, P3, C_SZ);
  gemm_bias_kernel<<<gGemm, 256, 0, stream>>>(x,  512, nullptr, 0, rgbv_w, rgbv_b, P4, C_SZ);
  gemm_bias_kernel<<<gGemm, 256, 0, stream>>>(x2, 512, nullptr, 0, infv_w, infv_b, P5, C_SZ);
  // 14. rgb branch: softmax(mask(aw * qK_inf/8)) @ rgb_v -> P0
  attn_kernel<<<gAttn, 128, 0, stream>>>(P1, P3, P4, aw, mask, nullptr, nullptr,
                                         nullptr, P0, 1);
  // 15. inf branch: softmax(mask(aw * qK_rgb/8)) @ inf_v -> P3 (infk now dead)
  attn_kernel<<<gAttn, 128, 0, stream>>>(P1, P2, P5, aw, mask, nullptr, nullptr,
                                         nullptr, P3, 1);
  // 16-17. output projections
  gemm_bias_kernel<<<gGemm, 256, 0, stream>>>(P0, 512, nullptr, 0, rgbo_w, rgbo_b, out_rgb, C_SZ);
  gemm_bias_kernel<<<gGemm, 256, 0, stream>>>(P3, 512, nullptr, 0, info_w, info_b, out_inf, C_SZ);
}
